// Weight_Vec_Ent_Sink_31525059953387
// MI455X (gfx1250) — matrix operand order repaired and checked by reading, not compiled, not run on hardware
//
#include <hip/hip_runtime.h>
#include <stdint.h>

// Sinkhorn (entropic OT) for MI455X / gfx1250.
// K = exp(-D/sigma) cached once in f16 (64 MB -> resident in 192 MB L2).
// Each iteration = two WMMA GEMVs (v_wmma_f32_16x16x32_f16, f32 accumulate).
// GEMV1 stages K tiles global->LDS with async-to-LDS b128 copies (ASYNCcnt),
// double-buffered, then gathers WMMA B-fragments from LDS.

#define NN 2048
#define MM 16384
#define SIGMA 0.1f
#define TOL 5e-4f
#define EPS 1e-7f
#define MAX_ITERS 100

typedef __attribute__((ext_vector_type(16))) _Float16 v16h;
typedef __attribute__((ext_vector_type(8)))  _Float16 v8h;
typedef __attribute__((ext_vector_type(4)))  _Float16 v4h;
typedef __attribute__((ext_vector_type(8)))  float    v8f;
typedef __attribute__((ext_vector_type(4)))  int      v4i;

// async-to-LDS builtin takes (v4i addrspace(1)*, v4i addrspace(3)*, i32, i32)
typedef __attribute__((address_space(1))) v4i* gv4i_p;
typedef __attribute__((address_space(3))) v4i* lv4i_p;

#if __has_builtin(__builtin_amdgcn_global_load_async_to_lds_b128)
#define HAVE_ASYNC_LDS 1
#else
#define HAVE_ASYNC_LDS 0
#endif

__device__ __forceinline__ void async_wait0() {
#if __has_builtin(__builtin_amdgcn_s_wait_asynccnt)
    __builtin_amdgcn_s_wait_asynccnt(0);
#else
    asm volatile("s_wait_asynccnt 0x0" ::: "memory");
#endif
}

__device__ __forceinline__ v16h cat8(v8h lo, v8h hi) {
    v16h r;
#pragma unroll
    for (int e = 0; e < 8; ++e) { r[e] = lo[e]; r[8 + e] = hi[e]; }
    return r;
}

// ---------------------------------------------------------------- precompute
__global__ void k_precompute(const float* __restrict__ D, _Float16* __restrict__ Kh) {
    size_t idx = ((size_t)blockIdx.x * blockDim.x + threadIdx.x) * 4;
    float4 d = *(const float4*)(D + idx);
    v4h h;
    h[0] = (_Float16)__expf(-d.x / SIGMA);
    h[1] = (_Float16)__expf(-d.y / SIGMA);
    h[2] = (_Float16)__expf(-d.z / SIGMA);
    h[3] = (_Float16)__expf(-d.w / SIGMA);
    *(v4h*)(Kh + idx) = h;
}

__global__ void k_init_state(float* __restrict__ g, int* __restrict__ done) {
    for (int i = threadIdx.x; i < NN; i += blockDim.x) g[i] = 0.0f;
    if (threadIdx.x == 0) *done = 0;
}

// ------------------------------------------------------------- row weights A
__global__ void k_weights_a(const float* __restrict__ tm, const float* __restrict__ g,
                            _Float16* __restrict__ wAh, float* __restrict__ diffsum,
                            const int* __restrict__ done) {
    if (*done) return;
    int i = blockIdx.x * blockDim.x + threadIdx.x;
    wAh[i] = (_Float16)(tm[i] * __expf(g[i]));
    if (i == 0) *diffsum = 0.0f;
}

// ---------------------------------------------------- tile staging (32x128)
// 256 threads, each moves two 16-byte rows-of-8: fully coalesced b128 traffic.
__device__ __forceinline__ void stage_tile(const _Float16* __restrict__ Kh,
                                           _Float16* dst, int kk, int j0, int tid) {
    const int r = tid >> 4;          // 0..15
    const int c = (tid & 15) << 3;   // 0,8,...,120
    const _Float16* s0 = Kh + (size_t)(kk + r) * MM + (j0 + c);
    const _Float16* s1 = Kh + (size_t)(kk + 16 + r) * MM + (j0 + c);
    _Float16* d0 = dst + r * 128 + c;
    _Float16* d1 = dst + (16 + r) * 128 + c;
#if HAVE_ASYNC_LDS
    __builtin_amdgcn_global_load_async_to_lds_b128(
        (gv4i_p)(uintptr_t)s0, (lv4i_p)(uint32_t)(uintptr_t)d0, 0, 0);
    __builtin_amdgcn_global_load_async_to_lds_b128(
        (gv4i_p)(uintptr_t)s1, (lv4i_p)(uint32_t)(uintptr_t)d1, 0, 0);
#else
    *(v8h*)d0 = *(const v8h*)s0;
    *(v8h*)d1 = *(const v8h*)s1;
#endif
}

// -------------------------------------------------------- GEMV1 (f update)
// s[j] = sum_i wA[i]*K[i,j] (reduce over i = WMMA K-dim).  Wave -> 16 cols.
// A = wA chunk broadcast to 16 rows (16-bit A 16x32 layout: lanes0-15
// e0..7->K0..7, e8..15->K16..23; lanes16-31 +8).  B gathered from the LDS
// tile in A's k order (lanes0-15 K0..7,K16..23; lanes16-31 +8).
__global__ void k_gemv1(const _Float16* __restrict__ Kh, const _Float16* __restrict__ wAh,
                        const float* __restrict__ rho, float* __restrict__ f,
                        _Float16* __restrict__ wBh, const int* __restrict__ done) {
    if (*done) return;
    __shared__ _Float16 tile[2][32][128];
    const int tid  = threadIdx.x;
    const int lane = tid & 31;
    const int wave = tid >> 5;
    const int colg = wave * 16 + (lane & 15);   // column within the 128-tile
    const int j0   = blockIdx.x * 128;
    const bool hi  = lane >= 16;
    const int aoff = hi ? 8 : 0;
    const int boff = hi ? 16 : 0;

    stage_tile(Kh, &tile[0][0][0], 0, j0, tid);
    v8f acc = {};
    int buf = 0;
    for (int kk = 0; kk < NN; kk += 32) {
#if HAVE_ASYNC_LDS
        async_wait0();                       // this wave's async stage done
#endif
        __syncthreads();                     // tile[buf] visible to all waves
        if (kk + 32 < NN)                    // prefetch next chunk into buf^1
            stage_tile(Kh, &tile[buf ^ 1][0][0], kk + 32, j0, tid);

        v8h a0 = *(const v8h*)(wAh + kk + aoff);
        v8h a1 = *(const v8h*)(wAh + kk + 16 + aoff);
        v16h a = cat8(a0, a1);
        v16h b;
#pragma unroll
        for (int e = 0; e < 16; ++e) b[e] = tile[buf][(e < 8 ? aoff + e : 16 + aoff + (e - 8))][colg];
        acc = __builtin_amdgcn_wmma_f32_16x16x32_f16(false, a, false, b,
                                                     (short)0, acc, false, false);
        buf ^= 1;
    }
    if (lane < 16) {
        const int col = j0 + colg;
        float s  = acc[0];                   // all 16 D-rows identical
        float fj = -__logf(s + EPS) / (1.0f + SIGMA);
        f[col]   = fj;
        wBh[col] = (_Float16)(rho[col] * __expf(fj));
    }
}

// -------------------------------------------------------- GEMV2 (g update)
// s2[i] = sum_j wB[j]*K[i,j] (reduce over j).  A = real row-major K tile
// (two contiguous b128 loads/lane), B = wB chunk broadcast (one 32B load).
__global__ void k_gemv2(const _Float16* __restrict__ Kh, const _Float16* __restrict__ wBh,
                        const float* __restrict__ g, float* __restrict__ g_new,
                        float* __restrict__ diffsum, const int* __restrict__ done) {
    if (*done) return;
    __shared__ float red[8][16];
    const int lane = threadIdx.x & 31;
    const int wave = threadIdx.x >> 5;
    const int i0   = (blockIdx.x * 8 + wave) * 16;
    const int row  = i0 + (lane & 15);
    const bool hi  = lane >= 16;
    const int aoff = hi ? 8 : 0;
    const _Float16* prow = Kh + (size_t)row * MM;
    v8f acc = {};
    for (int jj = 0; jj < MM; jj += 32) {
        v8h a0 = *(const v8h*)(prow + jj + (hi ? 16 : 0));
        v8h a1 = *(const v8h*)(prow + jj + (hi ? 16 : 0) + 8);
        v16h a = cat8(a0, a1);
        v16h b = *(const v16h*)(wBh + jj + (hi ? 16 : 0));
        acc = __builtin_amdgcn_wmma_f32_16x16x32_f16(false, a, false, b,
                                                     (short)0, acc, false, false);
    }
    // C/D layout: lane0 VGPR r -> (M=r, N=0); lane16 VGPR r -> (M=8+r, N=0).
    if (lane == 0) {
#pragma unroll
        for (int r = 0; r < 8; ++r) red[wave][r] = acc[r];
    } else if (lane == 16) {
#pragma unroll
        for (int r = 0; r < 8; ++r) red[wave][8 + r] = acc[r];
    }
    __syncthreads();
    if (lane < 16) {
        float s2 = red[wave][lane];
        float gi = -__logf(s2 + EPS);
        g_new[i0 + lane] = gi;
        atomicAdd(diffsum, fabsf(gi - g[i0 + lane]));
    }
}

// ----------------------------------------------------- per-iteration finish
__global__ void k_finish_iter(float* __restrict__ g, const float* __restrict__ g_new,
                              const float* __restrict__ diffsum, int* __restrict__ done) {
    if (*done) return;
    for (int i = threadIdx.x; i < NN; i += blockDim.x) g[i] = g_new[i];
    if (threadIdx.x == 0) {
        float diff = SIGMA * (*diffsum) / (float)NN;
        if (diff < TOL) *done = 1;
    }
}

// --------------------------------------------------------------- final xfrm
__global__ void k_output(const float* __restrict__ f, const float* __restrict__ g,
                         float* __restrict__ out) {
    int idx = blockIdx.x * blockDim.x + threadIdx.x;
    if (idx < MM)            out[idx] = 1.0f - __expf(-f[idx] * SIGMA);
    else if (idx < MM + NN)  out[idx] = g[idx - MM] * SIGMA;
}

extern "C" void kernel_launch(void* const* d_in, const int* in_sizes, int n_in,
                              void* d_out, int out_size, void* d_ws, size_t ws_size,
                              hipStream_t stream) {
    const float* D   = (const float*)d_in[0];   // (N, M)
    const float* rho = (const float*)d_in[1];   // (M,)  pre_density
    const float* tm  = (const float*)d_in[2];   // (N,)  target_mass

    uint8_t* base = (uint8_t*)d_ws;
    size_t off = 0;
    _Float16* Kh   = (_Float16*)(base + off); off += (size_t)NN * MM * sizeof(_Float16); // 64 MB
    float*    g    = (float*)   (base + off); off += NN * sizeof(float);
    float*    gn   = (float*)   (base + off); off += NN * sizeof(float);
    float*    f    = (float*)   (base + off); off += MM * sizeof(float);
    _Float16* wAh  = (_Float16*)(base + off); off += NN * sizeof(_Float16);
    _Float16* wBh  = (_Float16*)(base + off); off += MM * sizeof(_Float16);
    float* diffsum = (float*)   (base + off); off += 256;
    int*   done    = (int*)     (base + off); off += 256;
    (void)ws_size; (void)in_sizes; (void)n_in; (void)out_size;

    k_precompute<<<dim3((unsigned)(((size_t)NN * MM) / 1024)), dim3(256), 0, stream>>>(D, Kh);
    k_init_state<<<dim3(1), dim3(256), 0, stream>>>(g, done);

    for (int t = 0; t < MAX_ITERS; ++t) {
        k_weights_a  <<<dim3(NN / 256), dim3(256), 0, stream>>>(tm, g, wAh, diffsum, done);
        k_gemv1      <<<dim3(MM / 128), dim3(256), 0, stream>>>(Kh, wAh, rho, f, wBh, done);
        k_gemv2      <<<dim3(NN / 128), dim3(256), 0, stream>>>(Kh, wBh, g, gn, diffsum, done);
        k_finish_iter<<<dim3(1),        dim3(256), 0, stream>>>(g, gn, diffsum, done);
    }
    k_output<<<dim3((MM + NN) / 256), dim3(256), 0, stream>>>(f, g, (float*)d_out);
}
